// Temporal_Attention_17085379903626
// MI455X (gfx1250) — matrix operand order repaired and checked by reading, not compiled, not run on hardware
//
#include <hip/hip_runtime.h>
#include <hip/hip_bf16.h>
#include <stdint.h>

// ---------------------------------------------------------------------------
// MHA forward for MI455X (gfx1250, wave32, WMMA 16x16x32 f16 -> f32)
//   B=2, N=2048, C=1024, H=16, hd=64
// ---------------------------------------------------------------------------

typedef __attribute__((ext_vector_type(16))) _Float16 v16h;
typedef __attribute__((ext_vector_type(8)))  float    v8f;

union Frag16 {
  v16h  h;
  uint4 u[2];
};

#define DIMC   1024
#define HEADS  16
#define HDIM   64
#define BATCH  2
#define SEQ    2048
#define MTOT   (BATCH * SEQ)   // 4096 rows
#define QKVN   (3 * DIMC)      // 3072 qkv output features
#define ATT_SCALE 0.125f       // 64^-0.5

// Async global->LDS copy path (gfx1250, ASYNCcnt); guarded so the file
// compiles on toolchains without the builtin.
#if defined(__has_builtin)
#if __has_builtin(__builtin_amdgcn_global_load_async_to_lds_b128)
#define HAVE_ASYNC_LDS 1
#endif
#if __has_builtin(__builtin_amdgcn_s_wait_asynccnt)
#define HAVE_WAIT_ASYNC 1
#endif
#endif

// Parameter types per the hipcc diagnostic: the builtin wants
//   (v4i addrspace(1)*, v4i addrspace(3)*, imm, imm)
typedef int b128_t __attribute__((vector_size(16)));
typedef __attribute__((address_space(1))) b128_t* gptr128_t;
typedef __attribute__((address_space(3))) b128_t* lptr128_t;

__device__ __forceinline__ void async_copy_b128(const void* gsrc, void* ldst) {
#if defined(HAVE_ASYNC_LDS)
  __builtin_amdgcn_global_load_async_to_lds_b128(
      (gptr128_t)gsrc, (lptr128_t)ldst, 0, 0);
#else
  *(uint4*)ldst = *(const uint4*)gsrc;
#endif
}

__device__ __forceinline__ void wait_async_copies() {
#if defined(HAVE_ASYNC_LDS)
#if defined(HAVE_WAIT_ASYNC)
  __builtin_amdgcn_s_wait_asynccnt(0);
#else
  asm volatile("s_wait_asynccnt 0" ::: "memory");
#endif
#endif
}

// ---- fragment loaders matching CDNA5 WMMA VGPR layouts (CDNA5 ISA §7.12.2)
// A 16x32 f16: lane m(0-15) row m, K = k0+hi*8..+7 (v0-3) and +16.. (v4-7)
__device__ __forceinline__ v16h load_a_frag(const _Float16* base, int ld,
                                            int row0, int k0, int lane) {
  int m  = lane & 15;
  int hi = lane >> 4;
  const _Float16* p = base + (size_t)(row0 + m) * ld + k0 + hi * 8;
  Frag16 f;
  f.u[0] = *(const uint4*)(p);       // K kbase..kbase+7
  f.u[1] = *(const uint4*)(p + 16);  // K kbase+16..kbase+23
  return f.h;
}

// B 32x16 f16: lane n(0-15) col n, K = k0+hi*8..+7 (v0-3) and +16.. (v4-7).
// Source stored row-major [col][k] (contraction dim contiguous). Works for
// both global and LDS base pointers (LDS reads lower to ds_load_b128).
__device__ __forceinline__ v16h load_b_frag(const _Float16* base, int ld,
                                            int col0, int k0, int lane) {
  int n  = lane & 15;
  int hi = lane >> 4;
  const _Float16* p = base + (size_t)(col0 + n) * ld + k0 + hi * 8;
  Frag16 f;
  f.u[0] = *(const uint4*)(p);
  f.u[1] = *(const uint4*)(p + 16);
  return f.h;
}

__device__ __forceinline__ v8f wmma_f16(v16h a, v16h b, v8f c) {
  return __builtin_amdgcn_wmma_f32_16x16x32_f16(false, a, false, b,
                                                (short)0, c, false, false);
}

// ---------------------------------------------------------------------------
// Kernel 0: f32 -> f16 conversion (grid-stride)
// ---------------------------------------------------------------------------
__global__ void cvt_f32_f16_k(const float* __restrict__ in,
                              _Float16* __restrict__ out, int n) {
  int i = blockIdx.x * blockDim.x + threadIdx.x;
  int stride = gridDim.x * blockDim.x;
  for (; i < n; i += stride) out[i] = (_Float16)in[i];
}

// ---------------------------------------------------------------------------
// GEMM core: one wave computes a 32x64 strip, software-pipelined so the
// loads for K-step k+32 are in flight while the 8 WMMAs for step k execute.
// ---------------------------------------------------------------------------
__device__ __forceinline__ void gemm_tile_32x64(const _Float16* __restrict__ A,
                                                const _Float16* __restrict__ W,
                                                int mtile, int nbase, int lane,
                                                v8f acc[8]) {
  v16h a0 = load_a_frag(A, DIMC, mtile, 0, lane);
  v16h a1 = load_a_frag(A, DIMC, mtile + 16, 0, lane);
  v16h b0 = load_b_frag(W, DIMC, nbase + 0, 0, lane);
  v16h b1 = load_b_frag(W, DIMC, nbase + 16, 0, lane);
  v16h b2 = load_b_frag(W, DIMC, nbase + 32, 0, lane);
  v16h b3 = load_b_frag(W, DIMC, nbase + 48, 0, lane);

  for (int k0 = 32; k0 < DIMC; k0 += 32) {
    __builtin_prefetch(A + (size_t)(mtile + (lane & 15)) * DIMC + k0 + 32, 0, 1);
    v16h na0 = load_a_frag(A, DIMC, mtile, k0, lane);
    v16h na1 = load_a_frag(A, DIMC, mtile + 16, k0, lane);
    v16h nb0 = load_b_frag(W, DIMC, nbase + 0, k0, lane);
    v16h nb1 = load_b_frag(W, DIMC, nbase + 16, k0, lane);
    v16h nb2 = load_b_frag(W, DIMC, nbase + 32, k0, lane);
    v16h nb3 = load_b_frag(W, DIMC, nbase + 48, k0, lane);
    acc[0] = wmma_f16(a0, b0, acc[0]);
    acc[1] = wmma_f16(a0, b1, acc[1]);
    acc[2] = wmma_f16(a0, b2, acc[2]);
    acc[3] = wmma_f16(a0, b3, acc[3]);
    acc[4] = wmma_f16(a1, b0, acc[4]);
    acc[5] = wmma_f16(a1, b1, acc[5]);
    acc[6] = wmma_f16(a1, b2, acc[6]);
    acc[7] = wmma_f16(a1, b3, acc[7]);
    a0 = na0; a1 = na1;
    b0 = nb0; b1 = nb1; b2 = nb2; b3 = nb3;
  }
  acc[0] = wmma_f16(a0, b0, acc[0]);
  acc[1] = wmma_f16(a0, b1, acc[1]);
  acc[2] = wmma_f16(a0, b2, acc[2]);
  acc[3] = wmma_f16(a0, b3, acc[3]);
  acc[4] = wmma_f16(a1, b0, acc[4]);
  acc[5] = wmma_f16(a1, b1, acc[5]);
  acc[6] = wmma_f16(a1, b2, acc[6]);
  acc[7] = wmma_f16(a1, b3, acc[7]);
}

// ---------------------------------------------------------------------------
// Kernel 1: QKV GEMM.  qkv[m, d] = sum_k x[m,k] * w_qkv[d,k]
// A 64-col strip never crosses a q/k/v boundary (1024) nor a head boundary
// (64), and a 32-row tile never crosses a batch boundary (2048), so the
// output routing is wave-uniform: one scalar branch, branch-free lane stores.
//   Q,K : [B,H,SEQ,HDIM]   (d-contiguous, good for QK^T fragments)
//   Vt  : [B,H,HDIM,SEQ]   (seq-contiguous, good for P*V B-fragments)
// ---------------------------------------------------------------------------
__global__ __launch_bounds__(256) void qkv_gemm_k(
    const _Float16* __restrict__ x16, const _Float16* __restrict__ w16,
    _Float16* __restrict__ q16, _Float16* __restrict__ k16,
    _Float16* __restrict__ vt16) {
  int lane  = threadIdx.x & 31;
  int wave  = threadIdx.x >> 5;
  int mtile = blockIdx.y * 256 + wave * 32;
  int nbase = blockIdx.x * 64;

  v8f acc[8] = {};
  gemm_tile_32x64(x16, w16, mtile, nbase, lane, acc);

  int hi = lane >> 4, nn = lane & 15;
  // wave-uniform routing
  int which = nbase >> 10;            // 0=q 1=k 2=v
  int h     = (nbase & 1023) >> 6;    // head (uniform across the 64 cols)
  int bq    = mtile >> 11;            // batch (uniform across the 32 rows)
  int ns0   = mtile & 2047;           // seq position of row 0
  size_t bh = (size_t)(bq * HEADS + h);

  if (which == 2) {
    _Float16* dstv = vt16 + bh * HDIM * SEQ;  // [d][seq]
#pragma unroll
    for (int half = 0; half < 2; ++half) {
#pragma unroll
      for (int t = 0; t < 4; ++t) {
#pragma unroll
        for (int r = 0; r < 8; ++r) {
          int ns = ns0 + half * 16 + r + hi * 8;   // C layout: v[r] -> M=r(+8)
          int d  = t * 16 + nn;
          dstv[(size_t)d * SEQ + ns] = (_Float16)acc[half * 4 + t][r];
        }
      }
    }
  } else {
    _Float16* dst = (which == 0 ? q16 : k16) + (bh * SEQ + ns0) * HDIM;
#pragma unroll
    for (int half = 0; half < 2; ++half) {
#pragma unroll
      for (int t = 0; t < 4; ++t) {
#pragma unroll
        for (int r = 0; r < 8; ++r) {
          int rrow = half * 16 + r + hi * 8;
          int d    = t * 16 + nn;
          dst[(size_t)rrow * HDIM + d] = (_Float16)acc[half * 4 + t][r];
        }
      }
    }
  }
}

// ---------------------------------------------------------------------------
// Kernel 2: flash attention.  8 waves/block share one (b,h); each wave owns a
// 16-query tile.  Per 32-key step the block cooperatively stages the K tile
// (32x64, 4KB) and V tile (64x32, 4KB) into LDS once (async copies tracked by
// ASYNCcnt) -- 8x less L2 traffic than per-wave loads.  Then per wave:
// S = Q K^T (4 wmma from LDS), online softmax in registers, P transposed
// C-layout -> A-layout through a per-wave LDS slab, O += P V (4 wmma).
// ---------------------------------------------------------------------------
__global__ __launch_bounds__(256) void flash_attn_k(
    const _Float16* __restrict__ q16, const _Float16* __restrict__ k16,
    const _Float16* __restrict__ vt16, _Float16* __restrict__ attn16) {
  __shared__ _Float16 kbuf[32 * HDIM];    // [key][d]   4 KB
  __shared__ _Float16 vbuf[HDIM * 32];    // [d][key]   4 KB
  __shared__ _Float16 pbuf[8][16 * 32];   // per-wave P staging, 8 KB

  int tid  = threadIdx.x;
  int lane = tid & 31;
  int wave = tid >> 5;
  int bh   = blockIdx.y;  // 0..31
  int b = bh >> 4, h = bh & 15;
  int qrow0 = (blockIdx.x * 8 + wave) * 16;

  const _Float16* Q  = q16  + (size_t)bh * SEQ * HDIM;
  const _Float16* K  = k16  + (size_t)bh * SEQ * HDIM;
  const _Float16* Vt = vt16 + (size_t)bh * HDIM * SEQ;

  // Q fragments reused across all 64 key steps: load once.
  v16h qa0 = load_a_frag(Q, HDIM, qrow0, 0, lane);
  v16h qa1 = load_a_frag(Q, HDIM, qrow0, 32, lane);

  v8f acc[4] = {};
  float mrow[8], lrow[8];
#pragma unroll
  for (int r = 0; r < 8; ++r) { mrow[r] = -1e30f; lrow[r] = 0.0f; }

  int hi = lane >> 4;
  _Float16* pslab = pbuf[wave];
  int vrow = tid >> 2, vchunk = tid & 3;  // V staging: thread -> (d-row, 16B)

  for (int kb = 0; kb < SEQ; kb += 32) {
    __syncthreads();  // previous tile fully consumed
    // --- cooperative staging: K tile contiguous (32*64 f16), V tile rows.
    async_copy_b128((const uint4*)(K + (size_t)kb * HDIM) + tid,
                    (uint4*)kbuf + tid);
    async_copy_b128((const uint4*)(Vt + (size_t)vrow * SEQ + kb) + vchunk,
                    (uint4*)vbuf + tid);
    wait_async_copies();
    __syncthreads();  // staged tile visible to all waves

    // --- S = scale * (Q K^T) for two adjacent 16-key tiles, K from LDS
    v8f s0 = {}, s1 = {};
    s0 = wmma_f16(qa0, load_b_frag(kbuf, HDIM, 0, 0, lane), s0);
    s0 = wmma_f16(qa1, load_b_frag(kbuf, HDIM, 0, 32, lane), s0);
    s1 = wmma_f16(qa0, load_b_frag(kbuf, HDIM, 16, 0, lane), s1);
    s1 = wmma_f16(qa1, load_b_frag(kbuf, HDIM, 16, 32, lane), s1);

    // --- online softmax; rows split across 16-lane halves, xor masks 1..8
    // reduce exactly one row.
#pragma unroll
    for (int r = 0; r < 8; ++r) {
      float a0 = s0[r] * ATT_SCALE;
      float a1 = s1[r] * ATT_SCALE;
      float mx = fmaxf(a0, a1);
      mx = fmaxf(mx, __shfl_xor(mx, 1, 32));
      mx = fmaxf(mx, __shfl_xor(mx, 2, 32));
      mx = fmaxf(mx, __shfl_xor(mx, 4, 32));
      mx = fmaxf(mx, __shfl_xor(mx, 8, 32));
      float mnew = fmaxf(mrow[r], mx);
      float corr = __expf(mrow[r] - mnew);
      float p0 = __expf(a0 - mnew);
      float p1 = __expf(a1 - mnew);
      float rs = p0 + p1;
      rs += __shfl_xor(rs, 1, 32);
      rs += __shfl_xor(rs, 2, 32);
      rs += __shfl_xor(rs, 4, 32);
      rs += __shfl_xor(rs, 8, 32);
      lrow[r] = lrow[r] * corr + rs;
      mrow[r] = mnew;
      acc[0][r] *= corr; acc[1][r] *= corr;
      acc[2][r] *= corr; acc[3][r] *= corr;
      // stage P element (C-layout -> row-major 16x32 tile in LDS)
      int prow = r + hi * 8;
      int pcol = lane & 15;
      pslab[prow * 32 + pcol]      = (_Float16)p0;
      pslab[prow * 32 + pcol + 16] = (_Float16)p1;
    }
    asm volatile("s_wait_dscnt 0" ::: "memory");  // intra-wave LDS RAW

    // --- reload P as an A-fragment (16x32)
    Frag16 pf;
    {
      int m = lane & 15;
      const _Float16* pp = pslab + m * 32 + hi * 8;
      pf.u[0] = *(const uint4*)(pp);
      pf.u[1] = *(const uint4*)(pp + 16);
    }

    // --- O += P V, V fragments from LDS (d-major tile)
#pragma unroll
    for (int t = 0; t < 4; ++t) {
      v16h vf = load_b_frag(vbuf, 32, t * 16, 0, lane);
      acc[t] = wmma_f16(pf.h, vf, acc[t]);
    }
  }

  // --- normalize and write back into [B, N, C] (head-interleaved) as f16
  int nn = lane & 15;
#pragma unroll
  for (int t = 0; t < 4; ++t) {
#pragma unroll
    for (int r = 0; r < 8; ++r) {
      int nq = qrow0 + r + hi * 8;
      int d  = t * 16 + nn;
      float v = acc[t][r] / lrow[r];
      attn16[((size_t)(b * SEQ + nq)) * DIMC + h * HDIM + d] = (_Float16)v;
    }
  }
}

// ---------------------------------------------------------------------------
// Kernel 3: output projection.  out[m,n] = sum_k a[m,k] * w_proj[n,k] + bias
// ---------------------------------------------------------------------------
__global__ __launch_bounds__(256) void proj_gemm_k(
    const _Float16* __restrict__ a16, const _Float16* __restrict__ w16,
    const float* __restrict__ bias, float* __restrict__ out) {
  int lane  = threadIdx.x & 31;
  int wave  = threadIdx.x >> 5;
  int mtile = blockIdx.y * 256 + wave * 32;
  int nbase = blockIdx.x * 64;

  v8f acc[8] = {};
  gemm_tile_32x64(a16, w16, mtile, nbase, lane, acc);

  int hi = lane >> 4, nn = lane & 15;
#pragma unroll
  for (int half = 0; half < 2; ++half) {
#pragma unroll
    for (int t = 0; t < 4; ++t) {
#pragma unroll
      for (int r = 0; r < 8; ++r) {
        int row = mtile + half * 16 + r + hi * 8;
        int col = nbase + t * 16 + nn;
        out[(size_t)row * DIMC + col] = acc[half * 4 + t][r] + bias[col];
      }
    }
  }
}

// ---------------------------------------------------------------------------
extern "C" void kernel_launch(void* const* d_in, const int* in_sizes, int n_in,
                              void* d_out, int out_size, void* d_ws, size_t ws_size,
                              hipStream_t stream) {
  (void)in_sizes; (void)n_in; (void)out_size; (void)ws_size;
  const float* x      = (const float*)d_in[0];
  const float* w_qkv  = (const float*)d_in[1];
  const float* w_proj = (const float*)d_in[2];
  const float* b_proj = (const float*)d_in[3];
  float* out = (float*)d_out;

  // workspace layout (f16 elements), ~50 MB total
  _Float16* ws     = (_Float16*)d_ws;
  _Float16* x16    = ws;                                   // 4096*1024
  _Float16* wq16   = x16  + (size_t)MTOT * DIMC;           // 3072*1024
  _Float16* wp16   = wq16 + (size_t)QKVN * DIMC;           // 1024*1024
  _Float16* q16    = wp16 + (size_t)DIMC * DIMC;           // 2*16*2048*64
  _Float16* k16    = q16  + (size_t)BATCH * HEADS * SEQ * HDIM;
  _Float16* vt16   = k16  + (size_t)BATCH * HEADS * SEQ * HDIM;
  _Float16* attn16 = vt16 + (size_t)BATCH * HEADS * SEQ * HDIM;  // 4096*1024

  cvt_f32_f16_k<<<1024, 256, 0, stream>>>(x, x16, MTOT * DIMC);
  cvt_f32_f16_k<<<1024, 256, 0, stream>>>(w_qkv, wq16, QKVN * DIMC);
  cvt_f32_f16_k<<<1024, 256, 0, stream>>>(w_proj, wp16, DIMC * DIMC);

  qkv_gemm_k<<<dim3(QKVN / 64, MTOT / 256), 256, 0, stream>>>(x16, wq16,
                                                              q16, k16, vt16);

  flash_attn_k<<<dim3((SEQ / 16) / 8, BATCH * HEADS), 256, 0, stream>>>(
      q16, k16, vt16, attn16);

  proj_gemm_k<<<dim3(DIMC / 64, MTOT / 256), 256, 0, stream>>>(attn16, wp16,
                                                               b_proj, out);
}
